// S4Model_72404558676022
// MI455X (gfx1250) — matrix operand order repaired and checked by reading, not compiled, not run on hardware
//
#include <hip/hip_runtime.h>
#include <hip/hip_bf16.h>

#define BB 16
#define LL 2048
#define HH 256
#define NN 64
#define NL 4
#define BL (BB * LL)      // 32768 rows
#define IN_DIM 40
#define OUT_DIM 33
#define KENC 64           // padded encoder K

typedef __attribute__((ext_vector_type(16))) __bf16 v16bf;
typedef __attribute__((ext_vector_type(8)))  float  v8f;

#define LDS_STRIDE 40   // ushort elems per tile row: 32 data + 8 pad (80B, 16B-aligned)

__device__ __forceinline__ unsigned short f2bf(float f) {
  unsigned int u = __float_as_uint(f);
  u += 0x7FFFu + ((u >> 16) & 1u);          // round-to-nearest-even
  return (unsigned short)(u >> 16);
}

union FragU { uint4 q[2]; v16bf v; };
union Pack8 { unsigned short us[8]; uint4 q; };

// ---- CDNA5 async global->LDS copy (ASYNCcnt-tracked), per CDNA5 ISA ----
__device__ __forceinline__ void async_ld_b128(void* ldsptr, const void* gptr) {
  unsigned lds_off = (unsigned)(unsigned long long)ldsptr;   // flat aperture: addr[31:0] == LDS offset
  unsigned long long ga = (unsigned long long)gptr;
  asm volatile("global_load_async_to_lds_b128 %0, %1, off"
               :: "v"(lds_off), "v"(ga) : "memory");
}
__device__ __forceinline__ void wait_asynccnt0() {
  asm volatile("s_wait_asynccnt 0x0" ::: "memory");
}

// A fragment, 16x32 bf16 (ISA 7.12.2): lane m=lane&15; half=lane>>4
__device__ __forceinline__ v16bf load_frag_A(const unsigned short* lds, int mbase, int lane) {
  int mm = lane & 15, half = lane >> 4;
  const unsigned short* p = lds + (mbase + mm) * LDS_STRIDE + half * 8;
  FragU f;
  f.q[0] = *(const uint4*)(p);
  f.q[1] = *(const uint4*)(p + 16);
  return f.v;
}

// B fragment, 32x16 bf16: lane n=lane&15; half=lane>>4; elem s -> k=half*8+s (s<8), 16+half*8+(s-8) (s>=8)
__device__ __forceinline__ v16bf load_frag_B(const unsigned short* lds, int nbase, int lane) {
  int nn = lane & 15, half = lane >> 4;
  const unsigned short* p = lds + (nbase + nn) * LDS_STRIDE + half * 8;
  FragU f;
  f.q[0] = *(const uint4*)(p);
  f.q[1] = *(const uint4*)(p + 16);
  return f.v;
}

// Unguarded stage: 64 rows x 32 cols f32 -> bf16 LDS tile; 128 threads, 16 elems each.
__device__ __forceinline__ void stage64(unsigned short* lds, const float* __restrict__ g,
                                        int row0, int ld, int kt, int tid) {
  int r  = tid >> 1;              // 0..63
  int kc = (tid & 1) * 16;        // 0 or 16
  const float* gp = g + (long)(row0 + r) * ld + kt + kc;
  __builtin_prefetch(gp + 32, 0, 3);         // global_prefetch_b8: next k-tile
  float4 q0 = ((const float4*)gp)[0];
  float4 q1 = ((const float4*)gp)[1];
  float4 q2 = ((const float4*)gp)[2];
  float4 q3 = ((const float4*)gp)[3];
  Pack8 a, b;
  a.us[0] = f2bf(q0.x); a.us[1] = f2bf(q0.y); a.us[2] = f2bf(q0.z); a.us[3] = f2bf(q0.w);
  a.us[4] = f2bf(q1.x); a.us[5] = f2bf(q1.y); a.us[6] = f2bf(q1.z); a.us[7] = f2bf(q1.w);
  b.us[0] = f2bf(q2.x); b.us[1] = f2bf(q2.y); b.us[2] = f2bf(q2.z); b.us[3] = f2bf(q2.w);
  b.us[4] = f2bf(q3.x); b.us[5] = f2bf(q3.y); b.us[6] = f2bf(q3.z); b.us[7] = f2bf(q3.w);
  *(uint4*)(lds + r * LDS_STRIDE + kc)     = a.q;
  *(uint4*)(lds + r * LDS_STRIDE + kc + 8) = b.q;
}

// Zero-pad copy: dst[r, c] = (r<rs && c<cs) ? src[r, c] : 0
__global__ __launch_bounds__(256) void pad2d(const float* __restrict__ src, float* __restrict__ dst,
                                             int rs, int cs, int rd, int cd) {
  int idx = blockIdx.x * 256 + threadIdx.x;
  if (idx >= rd * cd) return;
  int r = idx / cd, c = idx - r * cd;
  dst[idx] = (r < rs && c < cs) ? src[r * cs + c] : 0.0f;
}

// --------- GEMM: C[m,n] = sum_k A[m,k]*W[n,k] + bias[n]; K mult of 32; W row-padded ---------
// 128 threads = 4 waves; wave (wm,wn) computes 32x32; block tile 64x64.
// NGUARD=true only for the decoder (Nact=33); otherwise fully branch-free.
template <bool NGUARD>
__global__ __launch_bounds__(128) void wmma_gemm_bias(
    const float* __restrict__ A, int K,
    const float* __restrict__ W, const float* __restrict__ bias,
    float* __restrict__ C, int ldc, int Nact)
{
  __shared__ __align__(16) unsigned short ldsA[64 * LDS_STRIDE];
  __shared__ __align__(16) unsigned short ldsW[64 * LDS_STRIDE];
  int tid = threadIdx.x, lane = tid & 31, w = tid >> 5;
  int wm = w & 1, wn = w >> 1;
  int m0 = blockIdx.x * 64, n0 = blockIdx.y * 64;
  v8f acc[2][2] = {};
  for (int kt = 0; kt < K; kt += 32) {
    __syncthreads();
    stage64(ldsA, A, m0, K, kt, tid);
    stage64(ldsW, W, n0, K, kt, tid);
    __syncthreads();
    v16bf a0 = load_frag_A(ldsA, wm * 32,      lane);
    v16bf a1 = load_frag_A(ldsA, wm * 32 + 16, lane);
    v16bf b;
    b = load_frag_B(ldsW, wn * 32, lane);
    acc[0][0] = __builtin_amdgcn_wmma_f32_16x16x32_bf16(false, a0, false, b, (short)0, acc[0][0], false, false);
    acc[1][0] = __builtin_amdgcn_wmma_f32_16x16x32_bf16(false, a1, false, b, (short)0, acc[1][0], false, false);
    b = load_frag_B(ldsW, wn * 32 + 16, lane);
    acc[0][1] = __builtin_amdgcn_wmma_f32_16x16x32_bf16(false, a0, false, b, (short)0, acc[0][1], false, false);
    acc[1][1] = __builtin_amdgcn_wmma_f32_16x16x32_bf16(false, a1, false, b, (short)0, acc[1][1], false, false);
  }
  // epilogue: bias hoisted, pointer-strided stores; guards only when NGUARD
  int nn = lane & 15, half = lane >> 4;
  int ncol0 = n0 + wn * 32 + nn;
  int ncol1 = ncol0 + 16;
  bool ok0 = !NGUARD || (ncol0 < Nact);
  bool ok1 = !NGUARD || (ncol1 < Nact);
  float bv0 = ok0 ? bias[ncol0] : 0.0f;
  float bv1 = ok1 ? bias[ncol1] : 0.0f;
  float* cp = C + (long)(m0 + wm * 32 + half * 8) * ldc;
  #pragma unroll
  for (int mi = 0; mi < 2; ++mi) {
    #pragma unroll
    for (int v = 0; v < 8; ++v) {
      float* row = cp + (long)(mi * 16 + v) * ldc;
      if (ok0) row[ncol0] = acc[mi][0][v] + bv0;
      if (ok1) row[ncol1] = acc[mi][1][v] + bv1;
    }
  }
}

// --------- fused GLU: H += (Y W1^T + b1) * sigmoid(Y W2^T + b2); all dims 256 ---------
__global__ __launch_bounds__(128) void wmma_glu(
    const float* __restrict__ Y,
    const float* __restrict__ W1, const float* __restrict__ b1,
    const float* __restrict__ W2, const float* __restrict__ b2,
    float* __restrict__ Hio)
{
  __shared__ __align__(16) unsigned short ldsA [64 * LDS_STRIDE];
  __shared__ __align__(16) unsigned short ldsB1[64 * LDS_STRIDE];
  __shared__ __align__(16) unsigned short ldsB2[64 * LDS_STRIDE];
  int tid = threadIdx.x, lane = tid & 31, w = tid >> 5;
  int wm = w & 1, wn = w >> 1;
  int m0 = blockIdx.x * 64, n0 = blockIdx.y * 64;
  v8f g[2][2] = {}, s[2][2] = {};
  for (int kt = 0; kt < HH; kt += 32) {
    __syncthreads();
    stage64(ldsA,  Y,  m0, HH, kt, tid);
    stage64(ldsB1, W1, n0, HH, kt, tid);
    stage64(ldsB2, W2, n0, HH, kt, tid);
    __syncthreads();
    v16bf a0 = load_frag_A(ldsA, wm * 32,      lane);
    v16bf a1 = load_frag_A(ldsA, wm * 32 + 16, lane);
    v16bf b;
    b = load_frag_B(ldsB1, wn * 32, lane);
    g[0][0] = __builtin_amdgcn_wmma_f32_16x16x32_bf16(false, a0, false, b, (short)0, g[0][0], false, false);
    g[1][0] = __builtin_amdgcn_wmma_f32_16x16x32_bf16(false, a1, false, b, (short)0, g[1][0], false, false);
    b = load_frag_B(ldsB1, wn * 32 + 16, lane);
    g[0][1] = __builtin_amdgcn_wmma_f32_16x16x32_bf16(false, a0, false, b, (short)0, g[0][1], false, false);
    g[1][1] = __builtin_amdgcn_wmma_f32_16x16x32_bf16(false, a1, false, b, (short)0, g[1][1], false, false);
    b = load_frag_B(ldsB2, wn * 32, lane);
    s[0][0] = __builtin_amdgcn_wmma_f32_16x16x32_bf16(false, a0, false, b, (short)0, s[0][0], false, false);
    s[1][0] = __builtin_amdgcn_wmma_f32_16x16x32_bf16(false, a1, false, b, (short)0, s[1][0], false, false);
    b = load_frag_B(ldsB2, wn * 32 + 16, lane);
    s[0][1] = __builtin_amdgcn_wmma_f32_16x16x32_bf16(false, a0, false, b, (short)0, s[0][1], false, false);
    s[1][1] = __builtin_amdgcn_wmma_f32_16x16x32_bf16(false, a1, false, b, (short)0, s[1][1], false, false);
  }
  int nn = lane & 15, half = lane >> 4;
  int ncol0 = n0 + wn * 32 + nn;
  int ncol1 = ncol0 + 16;
  float b1v0 = b1[ncol0], b1v1 = b1[ncol1];
  float b2v0 = b2[ncol0], b2v1 = b2[ncol1];
  float* hp = Hio + (long)(m0 + wm * 32 + half * 8) * HH;
  #pragma unroll
  for (int mi = 0; mi < 2; ++mi) {
    #pragma unroll
    for (int v = 0; v < 8; ++v) {
      float* row = hp + (long)(mi * 16 + v) * HH;
      float gv0 = g[mi][0][v] + b1v0;
      float sv0 = s[mi][0][v] + b2v0;
      row[ncol0] += gv0 * (1.0f / (1.0f + __expf(-sv0)));
      float gv1 = g[mi][1][v] + b1v1;
      float sv1 = s[mi][1][v] + b2v1;
      row[ncol1] += gv1 * (1.0f / (1.0f + __expf(-sv1)));
    }
  }
}

// ---------------- LayerNorm: one wave per 256-elem row ----------------
__global__ __launch_bounds__(256) void layernorm_kernel(
    const float* __restrict__ Hin, float* __restrict__ Z,
    const float* __restrict__ scale, const float* __restrict__ bias)
{
  int lane = threadIdx.x & 31, w = threadIdx.x >> 5;
  long row = (long)blockIdx.x * 8 + w;
  const float* p = Hin + row * HH + lane * 8;
  float4 q0 = *(const float4*)(p);
  float4 q1 = *(const float4*)(p + 4);
  float v[8] = {q0.x, q0.y, q0.z, q0.w, q1.x, q1.y, q1.z, q1.w};
  float s = 0.f, ss = 0.f;
  #pragma unroll
  for (int j = 0; j < 8; ++j) { s += v[j]; ss += v[j] * v[j]; }
  #pragma unroll
  for (int off = 16; off > 0; off >>= 1) {
    s  += __shfl_xor(s,  off, 32);
    ss += __shfl_xor(ss, off, 32);
  }
  float mu  = s * (1.0f / HH);
  float var = ss * (1.0f / HH) - mu * mu;
  float rs  = rsqrtf(var + 1e-5f);
  float o[8];
  #pragma unroll
  for (int j = 0; j < 8; ++j) {
    int hh = lane * 8 + j;
    o[j] = (v[j] - mu) * rs * scale[hh] + bias[hh];
  }
  float* zp = Z + row * HH + lane * 8;
  *(float4*)(zp)     = make_float4(o[0], o[1], o[2], o[3]);
  *(float4*)(zp + 4) = make_float4(o[4], o[5], o[6], o[7]);
}

// ---- S4D diagonal recurrence (== causal FFT conv) + D-skip + tanh-GELU ----
// grid (B, H/8), block 256 = 8 waves; wave w scans channel h0+w, lane owns 2 complex states.
// z tiles double-buffered through LDS via GLOBAL_LOAD_ASYNC_TO_LDS_B128 (ASYNCcnt).
__global__ __launch_bounds__(256) void s4d_scan_gelu(
    const float* __restrict__ Zin, float* __restrict__ Yout,
    const float* __restrict__ lre, const float* __restrict__ lim,
    const float* __restrict__ cre, const float* __restrict__ cim,
    const float* __restrict__ dvec, const float* __restrict__ lstep)
{
  __shared__ __align__(16) float ztile[2][256][12];   // 8 used cols + pad
  int tid = threadIdx.x, lane = tid & 31, w = tid >> 5;
  int b  = blockIdx.x;
  int h0 = blockIdx.y * 8;
  int h  = h0 + w;

  float dt = __expf(lstep[h]);
  float dAre[2], dAim[2], dBre[2], dBim[2], Cre[2], Cim[2], sre[2], sim[2];
  #pragma unroll
  for (int j = 0; j < 2; ++j) {
    int n = 2 * lane + j;
    float lr = lre[h * NN + n], li = lim[h * NN + n];
    float e    = __expf(lr * dt);
    float dre  = e * __cosf(li * dt);
    float dim_ = e * __sinf(li * dt);
    dAre[j] = dre; dAim[j] = dim_;
    float inv = 1.0f / (lr * lr + li * li);        // dB = (dA-1)/lambda
    float nre = dre - 1.0f, nim = dim_;
    dBre[j] = (nre * lr + nim * li) * inv;
    dBim[j] = (nim * lr - nre * li) * inv;
    Cre[j] = cre[h * NN + n]; Cim[j] = cim[h * NN + n];
    sre[j] = 0.f; sim[j] = 0.f;
  }
  float dd = dvec[h];

  const float* zb = Zin  + (long)b * LL * HH + h0;
  float*       yb = Yout + (long)b * LL * HH + h0;

  // prologue: async-issue chunk 0
  {
    const float* gp = zb + (long)tid * HH;
    async_ld_b128(&ztile[0][tid][0], gp);
    async_ld_b128(&ztile[0][tid][4], gp + 4);
  }

  for (int c = 0; c < LL / 256; ++c) {
    int cur = c & 1;
    wait_asynccnt0();          // this wave's async loads of chunk c landed in LDS
    __syncthreads();           // everyone's landed
    if (c + 1 < LL / 256) {    // overlap next chunk's fetch with this chunk's scan
      const float* gp = zb + (long)((c + 1) * 256 + tid) * HH;
      async_ld_b128(&ztile[cur ^ 1][tid][0], gp);
      async_ld_b128(&ztile[cur ^ 1][tid][4], gp + 4);
    }
    for (int l = 0; l < 256; ++l) {
      float zv = ztile[cur][l][w];                 // LDS broadcast
      float part = 0.f;
      #pragma unroll
      for (int j = 0; j < 2; ++j) {
        float nr = dAre[j] * sre[j] - dAim[j] * sim[j] + dBre[j] * zv;
        float ni = dAre[j] * sim[j] + dAim[j] * sre[j] + dBim[j] * zv;
        sre[j] = nr; sim[j] = ni;
        part += Cre[j] * nr - Cim[j] * ni;         // Re(C * s)
      }
      #pragma unroll
      for (int off = 16; off > 0; off >>= 1)
        part += __shfl_xor(part, off, 32);
      float yv = 2.0f * part + dd * zv;
      float t = 0.7978845608028654f * (yv + 0.044715f * yv * yv * yv);
      float gelu = 0.5f * yv * (1.0f + tanhf(t));  // tanh-approx GELU (jax default)
      if (lane == 0) ztile[cur][l][w] = gelu;      // wave-private column, safe in place
    }
    __syncthreads();
    { // coalesced write-back of chunk c
      float4 q0 = *(float4*)&ztile[cur][tid][0];
      float4 q1 = *(float4*)&ztile[cur][tid][4];
      float4* p = (float4*)(yb + (long)(c * 256 + tid) * HH);
      p[0] = q0; p[1] = q1;
    }
  }
}

extern "C" void kernel_launch(void* const* d_in, const int* in_sizes, int n_in,
                              void* d_out, int out_size, void* d_ws, size_t ws_size,
                              hipStream_t stream) {
  (void)in_sizes; (void)n_in; (void)out_size; (void)ws_size;
  const float* x        = (const float*)d_in[0];
  const float* enc_w    = (const float*)d_in[1];
  const float* enc_b    = (const float*)d_in[2];
  const float* lam_re   = (const float*)d_in[3];
  const float* lam_im   = (const float*)d_in[4];
  const float* c_re     = (const float*)d_in[5];
  const float* c_im     = (const float*)d_in[6];
  const float* dvec     = (const float*)d_in[7];
  const float* log_step = (const float*)d_in[8];
  const float* ln_scale = (const float*)d_in[9];
  const float* ln_bias  = (const float*)d_in[10];
  const float* out_w    = (const float*)d_in[11];
  const float* out_b    = (const float*)d_in[12];
  const float* out2_w   = (const float*)d_in[13];
  const float* out2_b   = (const float*)d_in[14];
  const float* dec_w    = (const float*)d_in[15];
  const float* dec_b    = (const float*)d_in[16];
  float* out = (float*)d_out;

  float* hbuf  = (float*)d_ws;                       // BL*HH
  float* zbuf  = hbuf  + (size_t)BL * HH;
  float* ybuf  = zbuf  + (size_t)BL * HH;
  float* xpad  = ybuf  + (size_t)BL * HH;            // BL x 64
  float* encwp = xpad  + (size_t)BL * KENC;          // 256 x 64
  float* decwp = encwp + (size_t)HH * KENC;          // 64 x 256

  // zero-pad ragged operands so GEMM staging is branch-free
  pad2d<<<(BL * KENC + 255) / 256, 256, 0, stream>>>(x,     xpad,  BL,      IN_DIM, BL, KENC);
  pad2d<<<(HH * KENC + 255) / 256, 256, 0, stream>>>(enc_w, encwp, HH,      IN_DIM, HH, KENC);
  pad2d<<<(64 * HH   + 255) / 256, 256, 0, stream>>>(dec_w, decwp, OUT_DIM, HH,     64, HH);

  // encoder: h = x @ enc_w^T + enc_b  (branch-free epilogue)
  wmma_gemm_bias<false><<<dim3(BL / 64, HH / 64), 128, 0, stream>>>(
      xpad, KENC, encwp, enc_b, hbuf, HH, HH);

  for (int i = 0; i < NL; ++i) {
    layernorm_kernel<<<BL / 8, 256, 0, stream>>>(
        hbuf, zbuf, ln_scale + i * HH, ln_bias + i * HH);
    s4d_scan_gelu<<<dim3(BB, HH / 8), 256, 0, stream>>>(
        zbuf, ybuf,
        lam_re + (size_t)i * HH * NN, lam_im + (size_t)i * HH * NN,
        c_re   + (size_t)i * HH * NN, c_im   + (size_t)i * HH * NN,
        dvec + i * HH, log_step + i * HH);
    wmma_glu<<<dim3(BL / 64, HH / 64), 128, 0, stream>>>(
        ybuf,
        out_w  + (size_t)i * HH * HH, out_b  + i * HH,
        out2_w + (size_t)i * HH * HH, out2_b + i * HH,
        hbuf);
  }

  // decoder: out = h @ dec_w^T + dec_b   (N=33, store-guarded)
  wmma_gemm_bias<true><<<dim3(BL / 64, 1), 128, 0, stream>>>(
      hbuf, HH, decwp, dec_b, out, OUT_DIM, OUT_DIM);
}
